// TrEncoderStg1_42588895707676
// MI455X (gfx1250) — matrix operand order repaired and checked by reading, not compiled, not run on hardware
//
#include <hip/hip_runtime.h>
#include <math.h>

typedef __attribute__((ext_vector_type(16))) _Float16 v16h;
typedef __attribute__((ext_vector_type(8)))  float    v8f;
typedef __attribute__((ext_vector_type(4)))  int      i32x4;

#if __has_builtin(__builtin_amdgcn_global_load_async_to_lds_b128)
#define HAS_ASYNC_LDS 1
#else
#define HAS_ASYNC_LDS 0
#endif

namespace cfg {
constexpr int Bn = 2, CIN = 5, Himg = 160, Wimg = 160;
constexpr int D = 48, HEADS = 3, DH = 24, WS = 800, NW = 32, LRW = 30;
constexpr int N  = Himg * Wimg;   // 25600
constexpr int NT = Bn * N;        // 51200
constexpr int INNER = HEADS * DH; // 72
constexpr int KT = 32;            // keys per tile
constexpr int NTILE = WS / KT;    // 25
}

__device__ __forceinline__ void wait_async_le1() {
#if HAS_ASYNC_LDS
#if __has_builtin(__builtin_amdgcn_s_wait_asynccnt)
  __builtin_amdgcn_s_wait_asynccnt(1);
#else
  asm volatile("s_wait_asynccnt 0x1" ::: "memory");
#endif
#endif
}
__device__ __forceinline__ void wait_async_le0() {
#if HAS_ASYNC_LDS
#if __has_builtin(__builtin_amdgcn_s_wait_asynccnt)
  __builtin_amdgcn_s_wait_asynccnt(0);
#else
  asm volatile("s_wait_asynccnt 0x0" ::: "memory");
#endif
#endif
}

// DPP-based max reduction across each 16-lane row (VALU pipe, not DS pipe).
template <int CTRL>
__device__ __forceinline__ float dpp_max_step(float x) {
  int t = __builtin_amdgcn_update_dpp(0, __float_as_int(x), CTRL, 0xF, 0xF, true);
  return fmaxf(x, __int_as_float(t));
}
__device__ __forceinline__ float rowmax16(float x) {
  x = dpp_max_step<0xB1>(x);   // quad_perm [1,0,3,2]  : xor 1
  x = dpp_max_step<0x4E>(x);   // quad_perm [2,3,0,1]  : xor 2
  x = dpp_max_step<0x141>(x);  // row_half_mirror      : cross quads within 8
  x = dpp_max_step<0x140>(x);  // row_mirror           : cross 8-halves within 16
  return x;
}

// ---------------- diag: e_diag/f_diag = einsum(hdw,dw->hd) + b ----------------
__global__ void diag_kernel(const float* __restrict__ Em, const float* __restrict__ Fm,
                            const float* __restrict__ we, const float* __restrict__ be,
                            const float* __restrict__ wf, const float* __restrict__ bf,
                            float* __restrict__ ed, float* __restrict__ fd) {
  using namespace cfg;
  int i = blockIdx.x * blockDim.x + threadIdx.x;
  if (i >= HEADS * DH) return;
  int d = i % DH;
  float se = 0.f, sf = 0.f;
  for (int w = 0; w < LRW; ++w) {
    se += Em[(size_t)i * LRW + w] * we[d * LRW + w];
    sf += Fm[(size_t)i * LRW + w] * wf[d * LRW + w];
  }
  ed[i] = se + be[d];
  fd[i] = sf + bf[d];
}

// ---------------- conv3x3 same + PReLU -> E (NCHW) and seq (B,N,D) ----------------
__global__ void conv_prelu_kernel(const float* __restrict__ x, const float* __restrict__ wgt,
                                  const float* __restrict__ bias, const float* __restrict__ pa,
                                  float* __restrict__ E, float* __restrict__ seq) {
  using namespace cfg;
  int idx = blockIdx.x * blockDim.x + threadIdx.x;
  if (idx >= Bn * D * Himg * Wimg) return;
  int xw = idx % Wimg; int t = idx / Wimg;
  int yh = t % Himg;   t /= Himg;
  int d  = t % D;      int b = t / D;
  float acc = bias[d];
  const float a = pa[0];
  for (int ci = 0; ci < CIN; ++ci) {
    const float* xp = x + ((size_t)(b * CIN + ci) * Himg) * Wimg;
    const float* wp = wgt + ((size_t)(d * CIN + ci)) * 9;
    __builtin_prefetch(xp + (yh + 2) * Wimg, 0, 0);
#pragma unroll
    for (int ky = 0; ky < 3; ++ky) {
      int yy = yh + ky - 1;
      if (yy < 0 || yy >= Himg) continue;
#pragma unroll
      for (int kx = 0; kx < 3; ++kx) {
        int xx = xw + kx - 1;
        if (xx < 0 || xx >= Wimg) continue;
        acc += xp[yy * Wimg + xx] * wp[ky * 3 + kx];
      }
    }
  }
  float e = fmaxf(acc, 0.f) + a * fminf(acc, 0.f);
  E[idx] = e;
  seq[((size_t)b * N + yh * Wimg + xw) * D + d] = e;
}

// ---------------- LayerNorm over D=48 (one thread per token) ----------------
__global__ void ln_kernel(const float* __restrict__ in, const float* __restrict__ g,
                          const float* __restrict__ bta, float* __restrict__ out) {
  using namespace cfg;
  int tkn = blockIdx.x * blockDim.x + threadIdx.x;
  if (tkn >= NT) return;
  const float* p = in + (size_t)tkn * D;
  float mu = 0.f, s2 = 0.f;
#pragma unroll 8
  for (int i = 0; i < D; ++i) { float v = p[i]; mu += v; s2 += v * v; }
  mu /= D; s2 = s2 / D - mu * mu;
  float r = rsqrtf(s2 + 1e-5f);
  float* o = out + (size_t)tkn * D;
#pragma unroll 8
  for (int i = 0; i < D; ++i) o[i] = (p[i] - mu) * r * g[i] + bta[i];
}

// ---------------- qkv = LN(seq) x w_qkv^T; fold scale/e_diag/f_diag; emit f16 ----------------
__global__ void qkv_kernel(const float* __restrict__ xn, const float* __restrict__ wqkv,
                           const float* __restrict__ ed, const float* __restrict__ fd,
                           _Float16* __restrict__ Q, _Float16* __restrict__ K,
                           _Float16* __restrict__ V) {
  using namespace cfg;
  size_t idx = (size_t)blockIdx.x * blockDim.x + threadIdx.x;
  if (idx >= (size_t)NT * 3 * INNER) return;
  int o = (int)(idx % (3 * INNER));
  size_t tkn = idx / (3 * INNER);
  int h = o / (3 * DH);
  int c = o % (3 * DH);
  int which = c / DH;
  int d = c % DH;
  const float* xp = xn + tkn * D;
  const float* wp = wqkv + (size_t)o * D;
  float acc = 0.f;
#pragma unroll 8
  for (int i = 0; i < D; ++i) acc += xp[i] * wp[i];
  int b = (int)(tkn / N), n = (int)(tkn % N);
  size_t dst = ((size_t)(b * HEADS + h) * N + n) * DH + d;
  if (which == 0)      Q[dst] = (_Float16)(acc * 0.20412414523193153f); // 1/sqrt(24)
  else if (which == 1) K[dst] = (_Float16)(acc * ed[h * DH + d]);
  else                 V[dst] = (_Float16)(acc * fd[h * DH + d]);
}

// ---------------- flash-style windowed attention ----------------
// Triple-buffered async global->LDS K/V tiles, WMMA f32_16x16x32_f16,
// ones-column trick folds the softmax denominator into the PV WMMA,
// DPP row-max keeps reductions off the DS pipe.
__global__ __launch_bounds__(320)
void attn_kernel(const _Float16* __restrict__ Qg, const _Float16* __restrict__ Kg,
                 const _Float16* __restrict__ Vg, float* __restrict__ ow) {
  using namespace cfg;
  __shared__ _Float16 sK[3][32][32];   // [buf][key][dim] dims 24..31 zero
  __shared__ _Float16 sV[3][32][32];   // dim 24 column = 1.0 (row-sum trick)
  __shared__ _Float16 sP[10][16][32];  // per-wave P staging (C-layout -> A-layout)

  int blk = blockIdx.x;
  int qt = blk % 5;      blk /= 5;
  int h  = blk % HEADS;  blk /= HEADS;
  int w  = blk % NW;     blk /= NW;
  int b  = blk;

  const int tid  = threadIdx.x;
  const int wid  = tid >> 5;
  const int lane = tid & 31;
  const int ln16 = lane & 15;
  const bool hi  = lane >= 16;

  const int win0  = w * WS;
  const int qrow0 = qt * 160 + wid * 16;
  const size_t headBase = (size_t)(b * HEADS + h) * N * DH;

  // init pad columns of all 3 buffers: sK[.][.][24..31]=0, sV[.][.][24]=1, sV[.][.][25..31]=0
  for (int i = tid; i < 3 * 32 * 8; i += blockDim.x) {
    int buf = i / 256, r = (i >> 3) & 31, c = 24 + (i & 7);
    sK[buf][r][c] = (_Float16)0.f;
    sV[buf][r][c] = (c == 24) ? (_Float16)1.f : (_Float16)0.f;
  }

  // tile loader: 192 threads move 2x(32x24) f16 as 16B chunks
  auto issue_tile = [&](int t, int buf) {
    if (tid < 192) {
      int mat = tid / 96;
      int j   = tid % 96;
      int key = j / 3, c = j % 3;
      const _Float16* src = (mat ? Vg : Kg) + headBase +
                            (size_t)(win0 + t * KT + key) * DH + c * 8;
      _Float16* dst = (mat ? sV : sK)[buf][key] + c * 8;
#if HAS_ASYNC_LDS
      __builtin_amdgcn_global_load_async_to_lds_b128(
          (__attribute__((address_space(1))) i32x4*)(void*)src,
          (__attribute__((address_space(3))) i32x4*)(void*)dst, 0, 0);
#else
      *(float4*)dst = *(const float4*)src;
#endif
    }
  };

  // Q A-fragment (16x32, contiguous k = 16h + j), pre-scaled by 1/sqrt(24)
  v16h aq;
  {
    const _Float16* qp = Qg + headBase + (size_t)(win0 + qrow0 + ln16) * DH;
#pragma unroll
    for (int j = 0; j < 16; ++j) {
      int k = (hi ? 16 : 0) + j;
      aq[j] = (k < DH) ? qp[k] : (_Float16)0.f;
    }
  }

  float m[8];
  v8f o0 = {}, o1 = {};
#pragma unroll
  for (int r = 0; r < 8; ++r) m[r] = -1e30f;

  issue_tile(0, 0);

  for (int t = 0; t < NTILE; ++t) {
    const int buf = t % 3;
    if (t + 1 < NTILE) {
      issue_tile(t + 1, (t + 1) % 3);
      wait_async_le1();   // oldest (tile t) complete; tile t+1 may stay in flight
    } else {
      wait_async_le0();   // drain before final tile's compute
    }
    __syncthreads();

    // B fragments of K^T (32 dims x 16 keys each)
    v16h bk0, bk1;
    {
      int kb = hi ? 16 : 0;
#pragma unroll
      for (int j = 0; j < 16; ++j) {
        bk0[j] = sK[buf][ln16][kb + j];
        bk1[j] = sK[buf][16 + ln16][kb + j];
      }
    }
    v8f z = {};
    v8f s0 = __builtin_amdgcn_wmma_f32_16x16x32_f16(false, aq, false, bk0, (short)0, z, false, false);
    v8f s1 = __builtin_amdgcn_wmma_f32_16x16x32_f16(false, aq, false, bk1, (short)0, z, false, false);

    // online softmax: DPP row-max, exp; denominator handled by V ones-column
#pragma unroll
    for (int r = 0; r < 8; ++r) {
      float mx = rowmax16(fmaxf(s0[r], s1[r]));
      float mn = fmaxf(m[r], mx);
      float alpha = __expf(m[r] - mn);
      m[r] = mn;
      float p0 = __expf(s0[r] - mn);
      float p1 = __expf(s1[r] - mn);
      o0[r] *= alpha;
      o1[r] *= alpha;
      int prow = hi ? (r + 8) : r;
      sP[wid][prow][ln16]      = (_Float16)p0;
      sP[wid][prow][ln16 + 16] = (_Float16)p1;
    }
    asm volatile("s_wait_dscnt 0" ::: "memory");  // wave-local LDS RAW fence

    // P as A-fragment (16 rows x 32 keys)
    v16h ap;
#pragma unroll
    for (int j = 0; j < 16; ++j) {
      int k = (hi ? 16 : 0) + j;
      ap[j] = sP[wid][ln16][k];
    }
    // V as B fragments: dims 0..15 and 16..31 (dim 24 column of ones -> row sums in o1[n==8])
    v16h bv0, bv1;
    {
      int kb = hi ? 16 : 0;
#pragma unroll
      for (int j = 0; j < 16; ++j) {
        bv0[j] = sV[buf][kb + j][ln16];
        bv1[j] = sV[buf][kb + j][16 + ln16];
      }
    }
    o0 = __builtin_amdgcn_wmma_f32_16x16x32_f16(false, ap, false, bv0, (short)0, o0, false, false);
    o1 = __builtin_amdgcn_wmma_f32_16x16x32_f16(false, ap, false, bv1, (short)0, o1, false, false);
  }

  // softmax denominator lives in o1 column n==8 (dim 24, the ones column).
  // Broadcast it across the 16-lane row, normalize, and store.
#pragma unroll
  for (int r = 0; r < 8; ++r) {
    float lsum = __shfl(o1[r], (lane & 16) | 8);
    float inv  = 1.f / lsum;
    int row    = qrow0 + (hi ? (r + 8) : r);
    size_t base = ((size_t)b * N + win0 + row) * (size_t)INNER + h * DH;
    ow[base + ln16] = o0[r] * inv;
    if (ln16 < 8) ow[base + 16 + ln16] = o1[r] * inv;
  }
}

// ---------------- out-projection (72->48) + residual ----------------
__global__ void outproj_kernel(const float* __restrict__ ow, const float* __restrict__ wout,
                               const float* __restrict__ bout, const float* __restrict__ seq,
                               float* __restrict__ seq2) {
  using namespace cfg;
  size_t idx = (size_t)blockIdx.x * blockDim.x + threadIdx.x;
  if (idx >= (size_t)NT * D) return;
  int o = (int)(idx % D);
  size_t tkn = idx / D;
  const float* p  = ow + tkn * INNER;
  const float* wp = wout + (size_t)o * INNER;
  float acc = bout[o];
#pragma unroll 8
  for (int i = 0; i < INNER; ++i) acc += p[i] * wp[i];
  seq2[idx] = seq[idx] + acc;
}

// ---------------- FF stage 1: leaky(LN2 x w1 + b1) ----------------
__global__ void ff1_kernel(const float* __restrict__ yn, const float* __restrict__ w1,
                           const float* __restrict__ b1, float* __restrict__ h1) {
  using namespace cfg;
  size_t idx = (size_t)blockIdx.x * blockDim.x + threadIdx.x;
  if (idx >= (size_t)NT * D) return;
  int o = (int)(idx % D);
  size_t tkn = idx / D;
  const float* p  = yn + tkn * D;
  const float* wp = w1 + (size_t)o * D;
  float acc = b1[o];
#pragma unroll 8
  for (int i = 0; i < D; ++i) acc += p[i] * wp[i];
  h1[idx] = (acc >= 0.f) ? acc : 0.01f * acc;
}

// ---------------- FF stage 2 + residual -> att1 ----------------
__global__ void ff2_kernel(const float* __restrict__ h1, const float* __restrict__ w2,
                           const float* __restrict__ b2, const float* __restrict__ seq2,
                           float* __restrict__ att1) {
  using namespace cfg;
  size_t idx = (size_t)blockIdx.x * blockDim.x + threadIdx.x;
  if (idx >= (size_t)NT * D) return;
  int c = (int)(idx % D);
  size_t tkn = idx / D;
  const float* hp = h1 + tkn * D;
  const float* wp = w2 + (size_t)c * D;
  float acc = b2[c];
#pragma unroll 8
  for (int o = 0; o < D; ++o) acc += hp[o] * wp[o];
  att1[idx] = seq2[idx] + acc;
}

// ---------------- Highpass1 = E*hp + E ----------------
__global__ void highpass_kernel(const float* __restrict__ E, const float* __restrict__ att1,
                                float* __restrict__ out) {
  using namespace cfg;
  int idx = blockIdx.x * blockDim.x + threadIdx.x;
  if (idx >= Bn * D * Himg * Wimg) return;
  int xw = idx % Wimg; int t = idx / Wimg;
  int yh = t % Himg;   t /= Himg;
  int d  = t % D;      int b = t / D;
  float hp = att1[((size_t)b * N + yh * Wimg + xw) * D + d];
  float e  = E[idx];
  out[idx] = e * hp + e;
}

// ---------------- final lin: att1 x w_fin^T + b_fin (48->96) ----------------
__global__ void fin_kernel(const float* __restrict__ att1, const float* __restrict__ wfin,
                           const float* __restrict__ bfin, float* __restrict__ out) {
  using namespace cfg;
  size_t idx = (size_t)blockIdx.x * blockDim.x + threadIdx.x;
  if (idx >= (size_t)NT * 2 * D) return;
  int o = (int)(idx % (2 * D));
  size_t tkn = idx / (2 * D);
  const float* p  = att1 + tkn * D;
  const float* wp = wfin + (size_t)o * D;
  float acc = bfin[o];
#pragma unroll 8
  for (int i = 0; i < D; ++i) acc += p[i] * wp[i];
  out[idx] = acc;
}

extern "C" void kernel_launch(void* const* d_in, const int* in_sizes, int n_in,
                              void* d_out, int out_size, void* d_ws, size_t ws_size,
                              hipStream_t stream) {
  using namespace cfg;
  const float* x       = (const float*)d_in[0];
  const float* conv_w  = (const float*)d_in[1];
  const float* conv_b  = (const float*)d_in[2];
  const float* prelu_a = (const float*)d_in[3];
  const float* ln1_g   = (const float*)d_in[4];
  const float* ln1_b   = (const float*)d_in[5];
  const float* w_qkv   = (const float*)d_in[6];
  const float* E_mat   = (const float*)d_in[7];
  const float* F_mat   = (const float*)d_in[8];
  const float* w_e     = (const float*)d_in[9];
  const float* b_e     = (const float*)d_in[10];
  const float* w_f     = (const float*)d_in[11];
  const float* b_f     = (const float*)d_in[12];
  const float* w_out   = (const float*)d_in[13];
  const float* b_out   = (const float*)d_in[14];
  const float* ln2_g   = (const float*)d_in[15];
  const float* ln2_b   = (const float*)d_in[16];
  const float* ff_w1   = (const float*)d_in[17];
  const float* ff_b1   = (const float*)d_in[18];
  const float* ff_w2   = (const float*)d_in[19];
  const float* ff_b2   = (const float*)d_in[20];
  const float* w_fin   = (const float*)d_in[21];
  const float* b_fin   = (const float*)d_in[22];

  char* ws = (char*)d_ws;
  size_t off = 0;
  auto alloc_f = [&](size_t n) -> float* {
    float* p = (float*)(ws + off); off += ((n * 4 + 255) / 256) * 256; return p; };
  auto alloc_h = [&](size_t n) -> _Float16* {
    _Float16* p = (_Float16*)(ws + off); off += ((n * 2 + 255) / 256) * 256; return p; };

  const size_t NE = (size_t)Bn * D * Himg * Wimg;  // 2,457,600
  float*    Ebuf = alloc_f(NE);
  float*    seq  = alloc_f(NE);
  float*    xn   = alloc_f(NE);
  float*    seq2 = alloc_f(NE);
  float*    yn   = alloc_f(NE);
  float*    h1   = alloc_f(NE);
  float*    att1 = alloc_f(NE);
  float*    ow   = alloc_f((size_t)NT * INNER);
  _Float16* Qb   = alloc_h((size_t)Bn * HEADS * N * DH);
  _Float16* Kb   = alloc_h((size_t)Bn * HEADS * N * DH);
  _Float16* Vb   = alloc_h((size_t)Bn * HEADS * N * DH);
  float*    ed   = alloc_f(HEADS * DH);
  float*    fd   = alloc_f(HEADS * DH);
  (void)ws_size; (void)in_sizes; (void)n_in; (void)out_size;

  float* out_hp  = (float*)d_out;        // Highpass1: 2,457,600 f32
  float* out_lin = out_hp + NE;          // lin: 4,915,200 f32

  diag_kernel<<<1, 128, 0, stream>>>(E_mat, F_mat, w_e, b_e, w_f, b_f, ed, fd);
  conv_prelu_kernel<<<(unsigned)((NE + 255) / 256), 256, 0, stream>>>(
      x, conv_w, conv_b, prelu_a, Ebuf, seq);
  ln_kernel<<<(NT + 255) / 256, 256, 0, stream>>>(seq, ln1_g, ln1_b, xn);
  qkv_kernel<<<(unsigned)(((size_t)NT * 3 * INNER + 255) / 256), 256, 0, stream>>>(
      xn, w_qkv, ed, fd, Qb, Kb, Vb);
  attn_kernel<<<Bn * NW * HEADS * 5, 320, 0, stream>>>(Qb, Kb, Vb, ow);
  outproj_kernel<<<(unsigned)(((size_t)NT * D + 255) / 256), 256, 0, stream>>>(
      ow, w_out, b_out, seq, seq2);
  ln_kernel<<<(NT + 255) / 256, 256, 0, stream>>>(seq2, ln2_g, ln2_b, yn);
  ff1_kernel<<<(unsigned)(((size_t)NT * D + 255) / 256), 256, 0, stream>>>(yn, ff_w1, ff_b1, h1);
  ff2_kernel<<<(unsigned)(((size_t)NT * D + 255) / 256), 256, 0, stream>>>(h1, ff_w2, ff_b2, seq2, att1);
  highpass_kernel<<<(unsigned)((NE + 255) / 256), 256, 0, stream>>>(Ebuf, att1, out_hp);
  fin_kernel<<<(unsigned)(((size_t)NT * 2 * D + 255) / 256), 256, 0, stream>>>(
      att1, w_fin, b_fin, out_lin);
}
